// CausalSelfAttention_15178414424258
// MI455X (gfx1250) — matrix operand order repaired and checked by reading, not compiled, not run on hardware
//
#include <hip/hip_runtime.h>
#include <hip/hip_bf16.h>
#include <stdint.h>

// ---- problem constants (match reference) ----
#define HQ_     16
#define HK_     4
#define DHEAD   64
#define WINDOW_ 1024
#define GATECH  12
#define BB      2
#define TT      2048
#define EE      1024              // HQ*D
#define NTOK    (BB*TT)           // 4096
#define QKVF    ((HQ_ + 2*HK_)*DHEAD)   // 1536

typedef __attribute__((ext_vector_type(16))) __bf16         v16bf;
typedef __attribute__((ext_vector_type(8)))  float          v8f;
typedef __attribute__((ext_vector_type(4)))  float          f32x4;
typedef __attribute__((ext_vector_type(4)))  unsigned int   u32x4;
typedef __attribute__((ext_vector_type(4)))  unsigned short u16x4;

union FragU { v16bf v; u32x4 q[2]; };

__device__ __forceinline__ unsigned short f2bf(float f) {
  union { float f; unsigned u; } x; x.f = f;
  unsigned r = x.u + 0x7FFFu + ((x.u >> 16) & 1u);   // round-to-nearest-even
  return (unsigned short)(r >> 16);
}

// ---- CDNA5 async VMEM->LDS copy (16B per lane), tracked by ASYNCcnt ----
__device__ __forceinline__ unsigned lds_addr32(const void* p) {
  return (unsigned)(unsigned long long)
      (__attribute__((address_space(3))) const void*)p;
}
__device__ __forceinline__ void cp_async16(void* lds, const void* g) {
  asm volatile("global_load_async_to_lds_b128 %0, %1, off"
               :: "v"(lds_addr32(lds)), "v"(g) : "memory");
}
__device__ __forceinline__ void wait_async0() {
  asm volatile("s_wait_asynccnt 0" ::: "memory");
}
__device__ __forceinline__ void wait_ds0() {
  asm volatile("s_wait_dscnt 0" ::: "memory");
}

// A fragment (16x32 bf16, MxK): lane half h reads K = h*8+[0..7] and 16+h*8+[0..7]
__device__ __forceinline__ v16bf load_fragA(const unsigned short* row, int half) {
  FragU f;
  f.q[0] = *(const u32x4*)(row + half * 8);
  f.q[1] = *(const u32x4*)(row + 16 + half * 8);
  return f.v;
}
// B fragment (32x16 bf16, KxN): lane half h reads K = h*8+[0..7] and 16+h*8+[0..7]
__device__ __forceinline__ v16bf load_fragB(const unsigned short* row, int half) {
  FragU f;
  f.q[0] = *(const u32x4*)(row + half * 8);
  f.q[1] = *(const u32x4*)(row + 16 + half * 8);
  return f.v;
}

#define WMMA_BF16(a, b, c) \
  __builtin_amdgcn_wmma_f32_16x16x32_bf16(false, (a), false, (b), (short)0, (c), false, false)

// =====================================================================
// Kernel 0: one-shot f32 -> bf16 conversion (x, w_qkv, w_o)
// =====================================================================
__global__ __launch_bounds__(256)
void cvt_f32_bf16(const float* __restrict__ in, unsigned short* __restrict__ outp,
                  int n /* multiple of 8 */) {
  int i = (blockIdx.x * 256 + threadIdx.x) * 8;
  if (i >= n) return;
  f32x4 a = *(const f32x4*)(in + i);
  f32x4 b = *(const f32x4*)(in + i + 4);
  u16x4 lo = { f2bf(a.x), f2bf(a.y), f2bf(a.z), f2bf(a.w) };
  u16x4 hi = { f2bf(b.x), f2bf(b.y), f2bf(b.z), f2bf(b.w) };
  *(u16x4*)(outp + i)     = lo;
  *(u16x4*)(outp + i + 4) = hi;
}

// =====================================================================
// Kernel 1/4: C[M,N] = A[M,K] * W[N,K]^T  (bf16 in, f32 out, bf16 WMMA)
// Block 256 thr (8 waves); tile 256(M) x 64(N); K-chunk 32.
// Each wave: 2 A-frags x 4 B-frags = 8 WMMA per chunk (B reused 2x).
// Double-buffered async VMEM->LDS staging; fragment loads batched so the
// 8 WMMAs issue back-to-back after a single DS wait.
// =====================================================================
#define LDSTR 40   // padded row stride (bf16): 80B -> 16B aligned, bank-spread

__global__ __launch_bounds__(256)
void gemm_bf16(const unsigned short* __restrict__ A,
               const unsigned short* __restrict__ W,
               float* __restrict__ C, int M, int N, int K) {
  __shared__ __align__(16) unsigned short sA[2][256 * LDSTR];
  __shared__ __align__(16) unsigned short sB[2][64 * LDSTR];
  const int tid  = threadIdx.x;
  const int lane = tid & 31, wave = tid >> 5;
  const int lnm  = lane & 15, half = lane >> 4;
  const int m0 = blockIdx.y * 256, n0 = blockIdx.x * 64;

  v8f zero = {0.f,0.f,0.f,0.f,0.f,0.f,0.f,0.f};
  v8f acc[8];
  #pragma unroll
  for (int i = 0; i < 8; ++i) acc[i] = zero;

  const int nk = K >> 5;
  const int rb = tid >> 2, cb = (tid & 3) * 8;   // B staging slot (1/thread)

  // prologue: stage chunk 0 into buffer 0 (A: 1024 x 16B -> 4/thread)
  #pragma unroll
  for (int p = 0; p < 4; ++p) {
    int s = tid + p * 256, r = s >> 2, c = (s & 3) * 8;
    cp_async16(&sA[0][r * LDSTR + c], A + (size_t)(m0 + r) * K + c);
  }
  cp_async16(&sB[0][rb * LDSTR + cb], W + (size_t)(n0 + rb) * K + cb);

  for (int it = 0; it < nk; ++it) {
    const int cur = it & 1;
    wait_async0();        // my async writes for buffer `cur` have landed
    __syncthreads();      // publish; also fences previous reads of buffer cur^1
    if (it + 1 < nk) {    // prefetch next chunk into the other buffer
      const int kc = (it + 1) << 5;
      #pragma unroll
      for (int p = 0; p < 4; ++p) {
        int s = tid + p * 256, r = s >> 2, c = (s & 3) * 8;
        cp_async16(&sA[cur ^ 1][r * LDSTR + c], A + (size_t)(m0 + r) * K + kc + c);
      }
      cp_async16(&sB[cur ^ 1][rb * LDSTR + cb], W + (size_t)(n0 + rb) * K + kc + cb);
    }
    // batch all fragment loads, then 8 back-to-back WMMAs
    const unsigned short* baseA = &sA[cur][(wave * 32 + lnm) * LDSTR];
    v16bf a0 = load_fragA(baseA, half);
    v16bf a1 = load_fragA(baseA + 16 * LDSTR, half);
    v16bf b0 = load_fragB(&sB[cur][(lnm)      * LDSTR], half);
    v16bf b1 = load_fragB(&sB[cur][(16 + lnm) * LDSTR], half);
    v16bf b2 = load_fragB(&sB[cur][(32 + lnm) * LDSTR], half);
    v16bf b3 = load_fragB(&sB[cur][(48 + lnm) * LDSTR], half);
    acc[0] = WMMA_BF16(a0, b0, acc[0]);
    acc[1] = WMMA_BF16(a0, b1, acc[1]);
    acc[2] = WMMA_BF16(a0, b2, acc[2]);
    acc[3] = WMMA_BF16(a0, b3, acc[3]);
    acc[4] = WMMA_BF16(a1, b0, acc[4]);
    acc[5] = WMMA_BF16(a1, b1, acc[5]);
    acc[6] = WMMA_BF16(a1, b2, acc[6]);
    acc[7] = WMMA_BF16(a1, b3, acc[7]);
  }
  // C/D layout: row m = half*8 + r, col n = j*16 + lnm
  #pragma unroll
  for (int mi = 0; mi < 2; ++mi) {
    const int mw = m0 + wave * 32 + mi * 16 + half * 8;
    #pragma unroll
    for (int j = 0; j < 4; ++j) {
      int col = n0 + j * 16 + lnm;
      #pragma unroll
      for (int r = 0; r < 8; ++r)
        C[(size_t)(mw + r) * N + col] = acc[mi * 4 + j][r];
    }
  }
}

// =====================================================================
// Kernel 2: gate+value-embed, RoPE + RMSNorm, f32 -> bf16 q/k/v
// One wave per token (D/2 == 32 == wave lanes).
// =====================================================================
__global__ __launch_bounds__(256)
void prep_kernel(const float* __restrict__ qkv, const float* __restrict__ x,
                 const float* __restrict__ ve, const float* __restrict__ cosT,
                 const float* __restrict__ sinT, const float* __restrict__ wg,
                 unsigned short* __restrict__ qb, unsigned short* __restrict__ kbp,
                 unsigned short* __restrict__ vbp) {
  const int tid = threadIdx.x;
  const int lane = tid & 31, wave = tid >> 5;
  const int token = blockIdx.x * 8 + wave;
  const int t = token % TT;
  const float* base = qkv + (size_t)token * QKVF;
  const float c = cosT[t * 32 + lane], s = sinT[t * 32 + lane];

  float gate[HK_];
  #pragma unroll
  for (int hk = 0; hk < HK_; ++hk) {
    float a = 0.f;
    #pragma unroll
    for (int ch = 0; ch < GATECH; ++ch)
      a += x[(size_t)token * EE + ch] * wg[hk * GATECH + ch];
    gate[hk] = 3.f / (1.f + __expf(-a));
  }
  #pragma unroll
  for (int h = 0; h < HQ_; ++h) {           // q heads: rope + rms
    float x1 = base[h * 64 + lane], x2 = base[h * 64 + 32 + lane];
    float y1 = x1 * c - x2 * s, y2 = x1 * s + x2 * c;
    float ss = y1 * y1 + y2 * y2;
    for (int m = 1; m < 32; m <<= 1) ss += __shfl_xor(ss, m, 32);
    float inv = rsqrtf(ss * (1.f / 64.f) + 1e-8f);
    qb[(size_t)token * EE + h * 64 + lane]      = f2bf(y1 * inv);
    qb[(size_t)token * EE + h * 64 + 32 + lane] = f2bf(y2 * inv);
  }
  #pragma unroll
  for (int h = 0; h < HK_; ++h) {           // k heads: rope + rms
    float x1 = base[1024 + h * 64 + lane], x2 = base[1024 + h * 64 + 32 + lane];
    float y1 = x1 * c - x2 * s, y2 = x1 * s + x2 * c;
    float ss = y1 * y1 + y2 * y2;
    for (int m = 1; m < 32; m <<= 1) ss += __shfl_xor(ss, m, 32);
    float inv = rsqrtf(ss * (1.f / 64.f) + 1e-8f);
    kbp[(size_t)token * 256 + h * 64 + lane]      = f2bf(y1 * inv);
    kbp[(size_t)token * 256 + h * 64 + 32 + lane] = f2bf(y2 * inv);
  }
  #pragma unroll
  for (int h = 0; h < HK_; ++h) {           // v heads: + gate*ve
    size_t vo = (size_t)token * 256 + h * 64;
    float v1 = base[1280 + h * 64 + lane]      + gate[h] * ve[vo + lane];
    float v2 = base[1280 + h * 64 + 32 + lane] + gate[h] * ve[vo + 32 + lane];
    vbp[vo + lane]      = f2bf(v1);
    vbp[vo + 32 + lane] = f2bf(v2);
  }
}

// =====================================================================
// Kernel 3: sliding-window flash attention, bf16 WMMA, bf16 output.
// 128 thr (4 waves) per (b, h, 64-query block); keys chunked by 32.
// K tile staged via async VMEM->LDS; V staged transposed (manual).
// =====================================================================
#define KSTR 72   // K-tile LDS row stride (bf16): 144B = 16B aligned
#define VSTR 40   // V^T-tile / P-tile stride:      80B = 16B aligned

__global__ __launch_bounds__(128)
void flash_kernel(const unsigned short* __restrict__ qb,
                  const unsigned short* __restrict__ kbp,
                  const unsigned short* __restrict__ vbp,
                  unsigned short* __restrict__ ob) {
  __shared__ __align__(16) unsigned short sK[32 * KSTR];
  __shared__ __align__(16) unsigned short sV[64 * VSTR];      // V transposed: [d][key]
  __shared__ __align__(16) unsigned short sP[4 * 16 * VSTR];  // per-wave P tile

  const int tid  = threadIdx.x;
  const int lane = tid & 31, wave = tid >> 5;
  const int lnm  = lane & 15, half = lane >> 4;
  const int nqt = TT / 64;
  const int qt = blockIdx.x % nqt;
  const int h  = (blockIdx.x / nqt) % HQ_;
  const int b  = blockIdx.x / (nqt * HQ_);
  const int hk = h >> 2;                 // jnp.repeat(k, 4): head h -> h/4
  const int q0 = qt * 64;
  const int qw = q0 + wave * 16;

  // preload this wave's Q (16x64) as two A fragments
  const unsigned short* qrow = qb + ((size_t)(b * TT + qw + lnm) * HQ_ + h) * 64;
  const v16bf qf0 = load_fragA(qrow, half);       // d 0..31
  const v16bf qf1 = load_fragA(qrow + 32, half);  // d 32..63

  v8f zero = {0.f,0.f,0.f,0.f,0.f,0.f,0.f,0.f};
  v8f acc[4]; acc[0]=zero; acc[1]=zero; acc[2]=zero; acc[3]=zero;
  float rmax[8], rsum[8];
  #pragma unroll
  for (int r = 0; r < 8; ++r) { rmax[r] = -1e30f; rsum[r] = 0.f; }

  int kstart = q0 - WINDOW_; if (kstart < 0) kstart = 0;
  kstart &= ~31;

  // per-thread K staging slots: 32 keys x 128B = 256 x 16B, 128 threads x 2
  const int rk0 = tid >> 2,         ck0 = (tid & 3) * 8;
  const int rk1 = (tid + 128) >> 2, ck1 = ((tid + 128) & 3) * 8;

  for (int kb0 = kstart; kb0 <= q0 + 63; kb0 += 32) {
    __syncthreads();   // everyone done reading previous K/V tiles
    // stage K tile (32 keys x 64 d) via async VMEM->LDS
    cp_async16(&sK[rk0 * KSTR + ck0],
               kbp + ((size_t)(b * TT + kb0 + rk0) * HK_ + hk) * 64 + ck0);
    cp_async16(&sK[rk1 * KSTR + ck1],
               kbp + ((size_t)(b * TT + kb0 + rk1) * HK_ + hk) * 64 + ck1);
    // stage V transposed: sV[d][key]
    {
      int key = tid >> 2, dseg = (tid & 3) * 16;
      const unsigned short* vr =
          vbp + ((size_t)(b * TT + kb0 + key) * HK_ + hk) * 64 + dseg;
      unsigned short tmp[16];
      *(u32x4*)(tmp)     = *(const u32x4*)(vr);
      *(u32x4*)(tmp + 8) = *(const u32x4*)(vr + 8);
      #pragma unroll
      for (int i = 0; i < 16; ++i) sV[(dseg + i) * VSTR + key] = tmp[i];
    }
    wait_async0();
    __syncthreads();

    // S = Q K^T : batch the 4 K fragments, then 4 back-to-back WMMAs
    const unsigned short* kr0 = sK + lnm * KSTR;
    const unsigned short* kr1 = sK + (16 + lnm) * KSTR;
    v16bf k00 = load_fragB(kr0,      half);
    v16bf k01 = load_fragB(kr0 + 32, half);
    v16bf k10 = load_fragB(kr1,      half);
    v16bf k11 = load_fragB(kr1 + 32, half);
    v8f s0 = WMMA_BF16(qf0, k00, zero);
    v8f s1 = WMMA_BF16(qf0, k10, zero);
    s0 = WMMA_BF16(qf1, k01, s0);
    s1 = WMMA_BF16(qf1, k11, s1);

    // online softmax over the 32 fresh columns (stats replicated per 16-lane half)
    const int kn0 = kb0 + lnm, kn1 = kb0 + 16 + lnm;
    #pragma unroll
    for (int r = 0; r < 8; ++r) {
      const int qm = qw + half * 8 + r;
      bool msk0 = (kn0 > qm) || (kn0 < qm - WINDOW_);
      bool msk1 = (kn1 > qm) || (kn1 < qm - WINDOW_);
      float t0 = msk0 ? -1e30f : s0[r] * 0.125f;      // 1/sqrt(64)
      float t1 = msk1 ? -1e30f : s1[r] * 0.125f;
      float mx = fmaxf(t0, t1);
      for (int sh = 1; sh < 16; sh <<= 1) mx = fmaxf(mx, __shfl_xor(mx, sh, 32));
      float nm = fmaxf(rmax[r], mx);
      float corr = __expf(rmax[r] - nm);              // ==1 while nothing valid yet
      float e0 = msk0 ? 0.f : __expf(t0 - nm);        // explicit zero: no -1e30 poison
      float e1 = msk1 ? 0.f : __expf(t1 - nm);
      float ps = e0 + e1;
      for (int sh = 1; sh < 16; sh <<= 1) ps += __shfl_xor(ps, sh, 32);
      rsum[r] = rsum[r] * corr + ps;
      rmax[r] = nm;
      #pragma unroll
      for (int j = 0; j < 4; ++j) acc[j][r] *= corr;
      const int m = half * 8 + r;
      sP[(wave * 16 + m) * VSTR + lnm]      = f2bf(e0);
      sP[(wave * 16 + m) * VSTR + 16 + lnm] = f2bf(e1);
    }
    wait_ds0();   // P writes visible to this wave's fragment reads

    // O += P V : batch P + 4 V fragments, then 4 back-to-back WMMAs
    v16bf pf = load_fragA(sP + (wave * 16 + lnm) * VSTR, half);
    v16bf vf0 = load_fragB(sV + (lnm)      * VSTR, half);
    v16bf vf1 = load_fragB(sV + (16 + lnm) * VSTR, half);
    v16bf vf2 = load_fragB(sV + (32 + lnm) * VSTR, half);
    v16bf vf3 = load_fragB(sV + (48 + lnm) * VSTR, half);
    acc[0] = WMMA_BF16(pf, vf0, acc[0]);
    acc[1] = WMMA_BF16(pf, vf1, acc[1]);
    acc[2] = WMMA_BF16(pf, vf2, acc[2]);
    acc[3] = WMMA_BF16(pf, vf3, acc[3]);
  }

  // epilogue: attn out in bf16 (feeds the bf16 output-projection GEMM)
  #pragma unroll
  for (int j = 0; j < 4; ++j) {
    int d = j * 16 + lnm;
    #pragma unroll
    for (int r = 0; r < 8; ++r) {
      int q = qw + half * 8 + r;
      ob[((size_t)(b * TT + q) * HQ_ + h) * 64 + d] = f2bf(acc[j][r] / rsum[r]);
    }
  }
}

// =====================================================================
extern "C" void kernel_launch(void* const* d_in, const int* in_sizes, int n_in,
                              void* d_out, int out_size, void* d_ws, size_t ws_size,
                              hipStream_t stream) {
  const float* x     = (const float*)d_in[0];
  const float* ve    = (const float*)d_in[1];
  const float* cosT  = (const float*)d_in[2];
  const float* sinT  = (const float*)d_in[3];
  const float* w_qkv = (const float*)d_in[4];
  const float* w_g   = (const float*)d_in[5];
  const float* w_o   = (const float*)d_in[6];
  float* out = (float*)d_out;

  char* ws = (char*)d_ws;
  size_t off = 0;
  float*          qkv  = (float*)(ws + off);          off += (size_t)NTOK * QKVF * 4; // 24 MB
  unsigned short* qb   = (unsigned short*)(ws + off); off += (size_t)NTOK * EE * 2;   //  8 MB
  unsigned short* kbp  = (unsigned short*)(ws + off); off += (size_t)NTOK * 256 * 2;  //  2 MB
  unsigned short* vbp  = (unsigned short*)(ws + off); off += (size_t)NTOK * 256 * 2;  //  2 MB
  unsigned short* attb = (unsigned short*)(ws + off); off += (size_t)NTOK * EE * 2;   //  8 MB
  unsigned short* xb   = (unsigned short*)(ws + off); off += (size_t)NTOK * EE * 2;   //  8 MB
  unsigned short* wqb  = (unsigned short*)(ws + off); off += (size_t)QKVF * EE * 2;   //  3 MB
  unsigned short* wob  = (unsigned short*)(ws + off); off += (size_t)EE * EE * 2;     //  2 MB

  const int nx  = NTOK * EE, nwq = QKVF * EE, nwo = EE * EE;
  // 0) one-shot f32 -> bf16 conversions
  cvt_f32_bf16<<<(nx  / 8 + 255) / 256, 256, 0, stream>>>(x,     xb,  nx);
  cvt_f32_bf16<<<(nwq / 8 + 255) / 256, 256, 0, stream>>>(w_qkv, wqb, nwq);
  cvt_f32_bf16<<<(nwo / 8 + 255) / 256, 256, 0, stream>>>(w_o,   wob, nwo);
  // 1) QKV projection: (4096 x 1024) * (1536 x 1024)^T
  gemm_bf16<<<dim3(QKVF / 64, NTOK / 256), 256, 0, stream>>>(
      xb, wqb, qkv, NTOK, QKVF, EE);
  // 2) gate/ve + RoPE + RMSNorm -> bf16 q/k/v
  prep_kernel<<<NTOK / 8, 256, 0, stream>>>(
      qkv, x, ve, cosT, sinT, w_g, qb, kbp, vbp);
  // 3) sliding-window flash attention -> bf16
  flash_kernel<<<BB * HQ_ * (TT / 64), 128, 0, stream>>>(qb, kbp, vbp, attb);
  // 4) output projection: (4096 x 1024) * (1024 x 1024)^T
  gemm_bf16<<<dim3(EE / 64, NTOK / 256), 256, 0, stream>>>(
      attb, wob, out, NTOK, EE, EE);
}
